// binary_dense_16415365005589
// MI455X (gfx1250) — compile-verified, matrix operand order checked by reading, not run on hardware
//
#include <hip/hip_runtime.h>
#include <hip/hip_bf16.h>
#include <stdint.h>

typedef _Float16 v16h __attribute__((ext_vector_type(16)));
typedef _Float16 v8h  __attribute__((ext_vector_type(8)));
typedef _Float16 v4h  __attribute__((ext_vector_type(4)));
typedef float    v8f  __attribute__((ext_vector_type(8)));
typedef int      v4i  __attribute__((ext_vector_type(4)));

static constexpr int Mdim = 8192;
static constexpr int Kdim = 4096;
static constexpr int Ndim = 4096;

#define BM 128
#define BN 128

// ---------------- async global->LDS copy (gfx1250 ASYNCcnt path) ----------------
#if defined(__has_builtin)
#  if __has_builtin(__builtin_amdgcn_global_load_async_to_lds_b128)
#    define HAVE_ASYNC_COPY 1
#  endif
#endif
#ifndef HAVE_ASYNC_COPY
#  define HAVE_ASYNC_COPY 0
#endif

#if HAVE_ASYNC_COPY
typedef __attribute__((address_space(1))) v4i glb_v4i;   // global int4
typedef __attribute__((address_space(3))) v4i lds_v4i;   // LDS int4
__device__ __forceinline__ void copy16(const void* g, void* l) {
  __builtin_amdgcn_global_load_async_to_lds_b128((glb_v4i*)g, (lds_v4i*)l, 0, 0);
}
__device__ __forceinline__ void copy_wait() {
#  if __has_builtin(__builtin_amdgcn_s_wait_asynccnt)
  __builtin_amdgcn_s_wait_asynccnt(0);
#  else
  asm volatile("s_wait_asynccnt 0" ::: "memory");
#  endif
}
#else
__device__ __forceinline__ void copy16(const void* g, void* l) {
  *(v8h*)l = *(const v8h*)g;
}
__device__ __forceinline__ void copy_wait() {}
#endif

// ---------------- pre-pass 1: W fp32 -> sign f16, transposed to [N][K] ----------------
__global__ __launch_bounds__(256)
void bin_transpose_kernel(const float* __restrict__ W, _Float16* __restrict__ WbT) {
  __shared__ _Float16 t[64][72];     // 72-half row stride keeps 8B align, breaks conflicts
  const int tid = threadIdx.x;
  const int c4  = tid & 15;          // float4 column (16 per 64)
  const int r   = tid >> 4;          // 0..15
  const int k0  = blockIdx.y * 64;
  const int n0  = blockIdx.x * 64;
#pragma unroll
  for (int p = 0; p < 4; ++p) {
    const int k = p * 16 + r;
    const float4 v = *(const float4*)&W[(size_t)(k0 + k) * Ndim + n0 + c4 * 4];
    float f[4] = {v.x, v.y, v.z, v.w};
    v4h s;
#pragma unroll
    for (int j = 0; j < 4; ++j)
      s[j] = (f[j] > 0.0f) ? (_Float16)1.0f
           : ((f[j] < 0.0f) ? (_Float16)-1.0f : (_Float16)0.0f);
    *(v4h*)&t[k][c4 * 4] = s;
  }
  __syncthreads();
#pragma unroll
  for (int p = 0; p < 4; ++p) {
    const int n = p * 16 + r;
    v4h o;
#pragma unroll
    for (int j = 0; j < 4; ++j) o[j] = t[c4 * 4 + j][n];
    *(v4h*)&WbT[(size_t)(n0 + n) * Kdim + k0 + c4 * 4] = o;
  }
}

// ---------------- pre-pass 2: X fp32 -> f16 hi/lo split ----------------
__global__ __launch_bounds__(256)
void split_hilo_kernel(const float* __restrict__ X,
                       _Float16* __restrict__ Xhi, _Float16* __restrict__ Xlo) {
  const size_t i = ((size_t)blockIdx.x * 256 + threadIdx.x) * 8;
  const float4 v0 = *(const float4*)(X + i);
  const float4 v1 = *(const float4*)(X + i + 4);
  float f[8] = {v0.x, v0.y, v0.z, v0.w, v1.x, v1.y, v1.z, v1.w};
  v8h hi, lo;
#pragma unroll
  for (int j = 0; j < 8; ++j) {
    _Float16 h = (_Float16)f[j];
    hi[j] = h;
    lo[j] = (_Float16)(f[j] - (float)h);
  }
  *(v8h*)(Xhi + i) = hi;
  *(v8h*)(Xlo + i) = lo;
}

// ---------------- main GEMM: pure-copy staging + WMMA ----------------
__global__ __launch_bounds__(256)
void gemm_wmma_pre(const _Float16* __restrict__ Xhi,
                   const _Float16* __restrict__ Xlo,
                   const _Float16* __restrict__ WbT,
                   float* __restrict__ Out) {
  constexpr int BK  = 64;
  constexpr int LDT = BK + 8;        // 72 halves = 144 B row stride (16B-aligned chunks)
  __shared__ _Float16 Ah[BM * LDT];
  __shared__ _Float16 Al[BM * LDT];
  __shared__ _Float16 Bt[BN * LDT];

  const int tid  = threadIdx.x;
  const int lane = tid & 31;
  const int wid  = tid >> 5;         // 0..7
  const int wrow = wid >> 1;         // 0..3 -> M offset *32
  const int wcol = wid & 1;          // 0..1 -> N offset *64
  const int lm   = lane & 15;
  const int kh   = lane >> 4;

  const int m0 = blockIdx.y * BM;
  const int n0 = blockIdx.x * BN;

  v8f acc[2][4] = {};

  for (int k0 = 0; k0 < Kdim; k0 += BK) {
    // stage 3 x (128 rows x 64 halves) tiles, 16B per lane per issue
#pragma unroll
    for (int p = 0; p < 4; ++p) {
      const int c   = p * 256 + tid;
      const int row = c >> 3;        // 0..127
      const int col = (c & 7) * 8;   // 0,8,..,56 halves
      copy16(&Xhi[(size_t)(m0 + row) * Kdim + k0 + col], &Ah[row * LDT + col]);
      copy16(&Xlo[(size_t)(m0 + row) * Kdim + k0 + col], &Al[row * LDT + col]);
      copy16(&WbT[(size_t)(n0 + row) * Kdim + k0 + col], &Bt[row * LDT + col]);
    }
    if (k0 + BK < Kdim) {
      const int row = tid >> 1, col = (tid & 1) * 32;
      __builtin_prefetch(&Xhi[(size_t)(m0 + row) * Kdim + k0 + BK + col], 0, 3);
      __builtin_prefetch(&Xlo[(size_t)(m0 + row) * Kdim + k0 + BK + col], 0, 3);
      __builtin_prefetch(&WbT[(size_t)(n0 + row) * Kdim + k0 + BK + col], 0, 3);
    }
    copy_wait();
    __syncthreads();

#pragma unroll
    for (int ks = 0; ks < BK; ks += 32) {
      v16h ah[2], al[2];
#pragma unroll
      for (int mt = 0; mt < 2; ++mt) {
        const int r = (wrow * 32 + mt * 16 + lm) * LDT + ks;
        v8h a0 = *(const v8h*)&Ah[r + kh * 8];
        v8h a1 = *(const v8h*)&Ah[r + 16 + kh * 8];
        ah[mt] = __builtin_shufflevector(a0, a1, 0,1,2,3,4,5,6,7,8,9,10,11,12,13,14,15);
        v8h b0 = *(const v8h*)&Al[r + kh * 8];
        v8h b1 = *(const v8h*)&Al[r + 16 + kh * 8];
        al[mt] = __builtin_shufflevector(b0, b1, 0,1,2,3,4,5,6,7,8,9,10,11,12,13,14,15);
      }
#pragma unroll
      for (int nt = 0; nt < 4; ++nt) {
        const int rc = (wcol * 64 + nt * 16 + lm) * LDT + ks;
        v8h c0 = *(const v8h*)&Bt[rc + kh * 8];
        v8h c1 = *(const v8h*)&Bt[rc + 16 + kh * 8];
        v16h bf = __builtin_shufflevector(c0, c1, 0,1,2,3,4,5,6,7,8,9,10,11,12,13,14,15);
#pragma unroll
        for (int mt = 0; mt < 2; ++mt) {
          acc[mt][nt] = __builtin_amdgcn_wmma_f32_16x16x32_f16(
              false, ah[mt], false, bf, (short)0, acc[mt][nt], false, false);
          acc[mt][nt] = __builtin_amdgcn_wmma_f32_16x16x32_f16(
              false, al[mt], false, bf, (short)0, acc[mt][nt], false, false);
        }
      }
    }
    __syncthreads();
  }

#pragma unroll
  for (int mt = 0; mt < 2; ++mt)
#pragma unroll
    for (int nt = 0; nt < 4; ++nt) {
      const int mbase = m0 + wrow * 32 + mt * 16 + kh * 8;
      const int nbase = n0 + wcol * 64 + nt * 16 + lm;
#pragma unroll
      for (int rr = 0; rr < 8; ++rr)
        Out[(size_t)(mbase + rr) * Ndim + nbase] = acc[mt][nt][rr];
    }
}

// ---------------- fallback: fused convert-in-kernel GEMM (round-1 version) ----------------
__global__ __launch_bounds__(256)
void gemm_wmma_fused(const float* __restrict__ X, const float* __restrict__ W,
                     float* __restrict__ Out) {
  constexpr int BK = 32;
  constexpr int LD = BK + 8;
  __shared__ _Float16 As_hi[BM * LD];
  __shared__ _Float16 As_lo[BM * LD];
  __shared__ _Float16 Bs[BN * LD];

  const int tid  = threadIdx.x;
  const int lane = tid & 31;
  const int wid  = tid >> 5;
  const int wrow = wid >> 1;
  const int wcol = wid & 1;
  const int lm   = lane & 15;
  const int kh   = lane >> 4;
  const int m0 = blockIdx.y * BM;
  const int n0 = blockIdx.x * BN;

  v8f acc[2][4] = {};
  const int a_c4 = tid & 7, a_r = tid >> 3;
  const int b_c4 = tid & 31, b_r = tid >> 5;

  for (int k0 = 0; k0 < Kdim; k0 += BK) {
#pragma unroll
    for (int p = 0; p < 4; ++p) {
      const int row = p * 32 + a_r;
      const float4 v = *(const float4*)&X[(size_t)(m0 + row) * Kdim + k0 + a_c4 * 4];
      float f[4] = {v.x, v.y, v.z, v.w};
      v4h hi, lo;
#pragma unroll
      for (int j = 0; j < 4; ++j) {
        _Float16 h = (_Float16)f[j];
        hi[j] = h;
        lo[j] = (_Float16)(f[j] - (float)h);
      }
      *(v4h*)&As_hi[row * LD + a_c4 * 4] = hi;
      *(v4h*)&As_lo[row * LD + a_c4 * 4] = lo;
    }
#pragma unroll
    for (int p = 0; p < 4; ++p) {
      const int krow = p * 8 + b_r;
      const float4 v = *(const float4*)&W[(size_t)(k0 + krow) * Ndim + n0 + b_c4 * 4];
      float f[4] = {v.x, v.y, v.z, v.w};
#pragma unroll
      for (int j = 0; j < 4; ++j)
        Bs[(b_c4 * 4 + j) * LD + krow] =
            (f[j] > 0.0f) ? (_Float16)1.0f : ((f[j] < 0.0f) ? (_Float16)-1.0f : (_Float16)0.0f);
    }
    __syncthreads();

    v16h ah[2], al[2];
#pragma unroll
    for (int mt = 0; mt < 2; ++mt) {
      const int r = (wrow * 32 + mt * 16 + lm) * LD;
      v8h a0 = *(const v8h*)&As_hi[r + kh * 8];
      v8h a1 = *(const v8h*)&As_hi[r + 16 + kh * 8];
      ah[mt] = __builtin_shufflevector(a0, a1, 0,1,2,3,4,5,6,7,8,9,10,11,12,13,14,15);
      v8h b0 = *(const v8h*)&As_lo[r + kh * 8];
      v8h b1 = *(const v8h*)&As_lo[r + 16 + kh * 8];
      al[mt] = __builtin_shufflevector(b0, b1, 0,1,2,3,4,5,6,7,8,9,10,11,12,13,14,15);
    }
#pragma unroll
    for (int nt = 0; nt < 4; ++nt) {
      const int rc = (wcol * 64 + nt * 16 + lm) * LD;
      v8h c0 = *(const v8h*)&Bs[rc + kh * 8];
      v8h c1 = *(const v8h*)&Bs[rc + 16 + kh * 8];
      v16h bf = __builtin_shufflevector(c0, c1, 0,1,2,3,4,5,6,7,8,9,10,11,12,13,14,15);
#pragma unroll
      for (int mt = 0; mt < 2; ++mt) {
        acc[mt][nt] = __builtin_amdgcn_wmma_f32_16x16x32_f16(
            false, ah[mt], false, bf, (short)0, acc[mt][nt], false, false);
        acc[mt][nt] = __builtin_amdgcn_wmma_f32_16x16x32_f16(
            false, al[mt], false, bf, (short)0, acc[mt][nt], false, false);
      }
    }
    __syncthreads();
  }

#pragma unroll
  for (int mt = 0; mt < 2; ++mt)
#pragma unroll
    for (int nt = 0; nt < 4; ++nt) {
      const int mbase = m0 + wrow * 32 + mt * 16 + kh * 8;
      const int nbase = n0 + wcol * 64 + nt * 16 + lm;
#pragma unroll
      for (int rr = 0; rr < 8; ++rr)
        Out[(size_t)(mbase + rr) * Ndim + nbase] = acc[mt][nt][rr];
    }
}

extern "C" void kernel_launch(void* const* d_in, const int* in_sizes, int n_in,
                              void* d_out, int out_size, void* d_ws, size_t ws_size,
                              hipStream_t stream) {
  const float* x = (const float*)d_in[0];
  const float* w = (const float*)d_in[1];
  float* out = (float*)d_out;

  const size_t WBT_BYTES = (size_t)Ndim * Kdim * sizeof(_Float16);  // 32 MiB
  const size_t XSP_BYTES = (size_t)Mdim * Kdim * sizeof(_Float16);  // 64 MiB each

  dim3 grid(Ndim / BN, Mdim / BM);   // (32, 64)
  dim3 block(256);

  if (ws_size >= WBT_BYTES + 2 * XSP_BYTES) {
    _Float16* wbt = (_Float16*)d_ws;
    _Float16* xhi = (_Float16*)((char*)d_ws + WBT_BYTES);
    _Float16* xlo = (_Float16*)((char*)d_ws + WBT_BYTES + XSP_BYTES);

    bin_transpose_kernel<<<dim3(Ndim / 64, Kdim / 64), block, 0, stream>>>(w, wbt);
    split_hilo_kernel<<<(Mdim * (size_t)Kdim) / 8 / 256, block, 0, stream>>>(x, xhi, xlo);
    gemm_wmma_pre<<<grid, block, 0, stream>>>(xhi, xlo, wbt, out);
  } else {
    gemm_wmma_fused<<<grid, block, 0, stream>>>(x, w, out);
  }
}
